// MicrostepKimiDeltaAttention_57251914056042
// MI455X (gfx1250) — compile-verified, matrix operand order checked by reading, not run on hardware
//
#include <hip/hip_runtime.h>
#include <hip/hip_bf16.h>

// ---------------------------------------------------------------------------
// MicrostepKimiDeltaAttention forward for MI455X (gfx1250, wave32, WMMA)
//
// Pipeline:
//   1) convert x + all weights fp32 -> bf16 (RNE)
//   2) bf16 WMMA GEMMs (v_wmma_f32_16x16x32_bf16) for all projections,
//      LDS staging via GLOBAL_LOAD_ASYNC_TO_LDS_B128 (ASYNCcnt path)
//   3) depthwise causal conv(K=4)+SiLU
//   4) prep: sigmoid(beta), decay=exp(-A*softplus(.)), l2norm(q)*D^-.5, l2norm(k)
//   5) sequential delta-rule scan: one block per (b,h), S[128][128] in VGPRs
//      (one column per lane, 128 lanes), k/q/decay broadcast from LDS
//   6) RMS-norm + sigmoid gate epilogue -> bf16
//   7) final WMMA GEMM -> d_out (fp32)
// ---------------------------------------------------------------------------

typedef __attribute__((ext_vector_type(16))) __bf16       v16bf;
typedef __attribute__((ext_vector_type(8)))  float        v8f;
typedef __attribute__((ext_vector_type(4)))  unsigned int u32x4;

// exact pointee type expected by the async-LDS builtins (per hipcc diagnostic):
typedef int v4i_gcc __attribute__((__vector_size__(4 * sizeof(int))));
typedef __attribute__((address_space(1))) v4i_gcc* async_gptr;
typedef __attribute__((address_space(3))) v4i_gcc* async_lptr;

union FragCast { u32x4 u[2]; v16bf v; };   // 32 bytes

#if __has_builtin(__builtin_amdgcn_global_load_async_to_lds_b128)
#define HAVE_ASYNC_LDS 1
#endif

__device__ __forceinline__ unsigned short f32_to_bf16(float f) {
    union { float f; unsigned int u; } c; c.f = f;
    unsigned int u = c.u;
    unsigned int r = u + 0x7fffu + ((u >> 16) & 1u);   // RNE
    return (unsigned short)(r >> 16);
}

__device__ __forceinline__ float softplus_f(float x) {
    return (x > 20.f) ? x : log1pf(__expf(x));
}
__device__ __forceinline__ float sigmoid_f(float x) {
    return 1.f / (1.f + __expf(-x));
}

// ---------------------------------------------------------------------------
// fp32 -> bf16 conversion
// ---------------------------------------------------------------------------
__global__ void cvt_bf16_kernel(const float* __restrict__ in,
                                unsigned short* __restrict__ out, size_t n) {
    size_t i = (size_t)blockIdx.x * blockDim.x + threadIdx.x;
    if (i < n) out[i] = f32_to_bf16(in[i]);
}

// ---------------------------------------------------------------------------
// bf16 WMMA GEMM:  C[M,N] = A[M,K] * W[N,K]^T   (both row-major, fp32 out)
// block: 256 threads (8 waves). Block tile 128x128, wave tile 64x32
//   -> 4x2 = 8 accumulators of v_wmma_f32_16x16x32_bf16 per K-chunk.
// Staging: async global->LDS (ASYNCcnt) when the builtin exists.
// Requires: M % 128 == 0, K % 32 == 0. N is guarded.
// ---------------------------------------------------------------------------
#define GEMM_BM 128
#define GEMM_BK 32
#define GEMM_LDSS 40   // padded row stride (elems); 80B, multiple of 16B

__global__ __launch_bounds__(256)
void gemm_bf16_kernel(const unsigned short* __restrict__ A,
                      const unsigned short* __restrict__ W,
                      float* __restrict__ C, int M, int N, int K) {
    __shared__ unsigned short As[GEMM_BM * GEMM_LDSS];
    __shared__ unsigned short Bs[GEMM_BM * GEMM_LDSS];

    const int tid   = threadIdx.x;
    const int lane  = tid & 31;
    const int wave  = tid >> 5;
    const int wm    = wave >> 2;       // 0..1  (M direction, 64 rows each)
    const int wn    = wave & 3;        // 0..3  (N direction, 32 cols each)
    const int laneLo = lane & 15;
    const int laneHi = lane >> 4;      // 0/1
    const int c0     = laneHi * 8;     // K sub-offset per ISA bf16 layout

    const int blockM0 = blockIdx.y * GEMM_BM;
    const int blockN0 = blockIdx.x * GEMM_BM;

    const v8f vzero = {0.f,0.f,0.f,0.f,0.f,0.f,0.f,0.f};
    v8f acc[4][2];
    #pragma unroll
    for (int i = 0; i < 4; ++i)
        #pragma unroll
        for (int j = 0; j < 2; ++j) acc[i][j] = vzero;

    for (int k0 = 0; k0 < K; k0 += GEMM_BK) {
        // ---- stage 128x32 bf16 tiles of A and W into LDS (16B chunks) ----
#ifdef HAVE_ASYNC_LDS
        #pragma unroll
        for (int it = 0; it < 2; ++it) {
            int chunk = tid + it * 256;            // 0..511
            int row   = chunk >> 2;                // 0..127
            int col   = (chunk & 3) << 3;          // 0,8,16,24
            const unsigned short* gA = A + (size_t)(blockM0 + row) * K + k0 + col;
            __builtin_amdgcn_global_load_async_to_lds_b128(
                (async_gptr)gA, (async_lptr)&As[row * GEMM_LDSS + col], 0, 0);
            int nrow = blockN0 + row;
            if (nrow < N) {
                const unsigned short* gB = W + (size_t)nrow * K + k0 + col;
                __builtin_amdgcn_global_load_async_to_lds_b128(
                    (async_gptr)gB, (async_lptr)&Bs[row * GEMM_LDSS + col], 0, 0);
            } else {
                u32x4 z = {0u, 0u, 0u, 0u};
                *(u32x4*)(&Bs[row * GEMM_LDSS + col]) = z;
            }
        }
#if __has_builtin(__builtin_amdgcn_s_wait_asynccnt)
        __builtin_amdgcn_s_wait_asynccnt(0);
#else
        asm volatile("s_wait_asynccnt 0" ::: "memory");
#endif
#else   // synchronous fallback staging
        #pragma unroll
        for (int it = 0; it < 2; ++it) {
            int chunk = tid + it * 256;            // 0..511
            int row   = chunk >> 2;                // 0..127
            int col   = (chunk & 3) << 3;          // 0,8,16,24
            const u32x4* ga =
                (const u32x4*)(A + (size_t)(blockM0 + row) * K + k0 + col);
            *(u32x4*)(&As[row * GEMM_LDSS + col]) = *ga;
            int nrow = blockN0 + row;
            u32x4 bv = {0u, 0u, 0u, 0u};
            if (nrow < N)
                bv = *(const u32x4*)(W + (size_t)nrow * K + k0 + col);
            *(u32x4*)(&Bs[row * GEMM_LDSS + col]) = bv;
        }
#endif
        // hint next A tile into cache (global_prefetch_b8)
        if (k0 + GEMM_BK < K && tid < 128)
            __builtin_prefetch(A + (size_t)(blockM0 + tid) * K + k0 + GEMM_BK, 0, 0);
        __syncthreads();

        // ---- load fragments per ISA 16-bit A/B layout ----
        FragCast afr[4], bfr[2];
        #pragma unroll
        for (int i = 0; i < 4; ++i) {
            const unsigned short* p =
                &As[(wm * 64 + i * 16 + laneLo) * GEMM_LDSS + c0];
            afr[i].u[0] = *(const u32x4*)(p);        // K = c0 .. c0+7
            afr[i].u[1] = *(const u32x4*)(p + 16);   // K = c0+16 .. c0+23
        }
        #pragma unroll
        for (int j = 0; j < 2; ++j) {
            const unsigned short* p =
                &Bs[(wn * 32 + j * 16 + laneLo) * GEMM_LDSS + c0];
            bfr[j].u[0] = *(const u32x4*)(p);
            bfr[j].u[1] = *(const u32x4*)(p + 16);
        }

        // ---- 8 WMMAs ----
        #pragma unroll
        for (int i = 0; i < 4; ++i)
            #pragma unroll
            for (int j = 0; j < 2; ++j)
                acc[i][j] = __builtin_amdgcn_wmma_f32_16x16x32_bf16(
                    false, afr[i].v, false, bfr[j].v,
                    (short)0, acc[i][j], false, false);
        __syncthreads();
    }

    // ---- store per 16x16 f32 D layout (VGPR r: M=r / M=8+r) ----
    #pragma unroll
    for (int i = 0; i < 4; ++i) {
        int mbase = blockM0 + wm * 64 + i * 16 + laneHi * 8;
        #pragma unroll
        for (int j = 0; j < 2; ++j) {
            int n = blockN0 + wn * 32 + j * 16 + laneLo;
            if (n < N) {
                #pragma unroll
                for (int r = 0; r < 8; ++r)
                    C[(size_t)(mbase + r) * N + n] = acc[i][j][r];
            }
        }
    }
}

// ---------------------------------------------------------------------------
// depthwise causal conv (K=4) + SiLU over (B,T,C)
// ---------------------------------------------------------------------------
__global__ void conv_silu_kernel(const float* __restrict__ pre,
                                 const float* __restrict__ w,
                                 float* __restrict__ out,
                                 int T_, int C_, long long total) {
    long long idx = (long long)blockIdx.x * blockDim.x + threadIdx.x;
    if (idx >= total) return;
    int c = (int)(idx % C_);
    long long bt = idx / C_;
    int t = (int)(bt % T_);
    float acc = 0.f;
    #pragma unroll
    for (int i = 0; i < 4; ++i) {
        int tt = t - 3 + i;
        if (tt >= 0)
            acc = fmaf(pre[idx + (long long)(tt - t) * C_], w[c * 4 + i], acc);
    }
    out[idx] = acc * sigmoid_f(acc);   // SiLU
}

// ---------------------------------------------------------------------------
// prep kernels
// ---------------------------------------------------------------------------
__global__ void decay_fill_kernel(const float* __restrict__ A_log,
                                  const float* __restrict__ dt_bias,
                                  float* __restrict__ dfill, int n) {
    int i = blockIdx.x * blockDim.x + threadIdx.x;
    if (i >= n) return;
    int h = i >> 7;                         // channel / 128
    float A = __expf(A_log[h]);
    dfill[i] = __expf(-A * softplus_f(-10000.f + dt_bias[i]));
}

__global__ void decay0_kernel(float* __restrict__ graw,
                              const float* __restrict__ A_log,
                              const float* __restrict__ dt_bias, long long n) {
    long long i = (long long)blockIdx.x * blockDim.x + threadIdx.x;
    if (i >= n) return;
    int ch = (int)(i & 2047);
    int h  = ch >> 7;
    float A = __expf(A_log[h]);
    graw[i] = __expf(-A * softplus_f(graw[i] + dt_bias[ch]));
}

__global__ void sigmoid_inplace_kernel(float* __restrict__ x, long long n) {
    long long i = (long long)blockIdx.x * blockDim.x + threadIdx.x;
    if (i < n) x[i] = sigmoid_f(x[i]);
}

// l2-normalize groups of 128 contiguous floats; one wave32 per group.
__global__ __launch_bounds__(256)
void l2norm_kernel(float* __restrict__ x, int groups, float extraScale) {
    int wave = threadIdx.x >> 5;
    int lane = threadIdx.x & 31;
    int g = blockIdx.x * 8 + wave;
    if (g >= groups) return;
    float* p = x + (size_t)g * 128;
    float v0 = p[lane], v1 = p[lane + 32], v2 = p[lane + 64], v3 = p[lane + 96];
    float s = v0 * v0 + v1 * v1 + v2 * v2 + v3 * v3;
    #pragma unroll
    for (int off = 16; off > 0; off >>= 1) s += __shfl_xor(s, off, 32);
    float scale = rsqrtf(s + 1e-6f) * extraScale;
    p[lane]      = v0 * scale;
    p[lane + 32] = v1 * scale;
    p[lane + 64] = v2 * scale;
    p[lane + 96] = v3 * scale;
}

// ---------------------------------------------------------------------------
// sequential delta-rule scan: one block (128 threads) per (b,h).
// Lane owns column v of state S[d][v]; S column lives in 128 VGPRs.
// ---------------------------------------------------------------------------
__global__ __launch_bounds__(128)
void scan_kernel(const float* __restrict__ q, const float* __restrict__ k,
                 const float* __restrict__ v, const float* __restrict__ beta,
                 const float* __restrict__ decay0,
                 const float* __restrict__ dfill,
                 const float* __restrict__ micro_logits,
                 float* __restrict__ o_mix) {
    const int bh = blockIdx.x;
    const int b  = bh >> 4;
    const int h  = bh & 15;
    const int vc = threadIdx.x;               // column of S, 0..127

    __shared__ float dfill_s[128];
    __shared__ float q_s[128];
    __shared__ float dec_s[128];
    __shared__ float k_s[4][128];
    __shared__ float beta_s[4];

    dfill_s[vc] = dfill[h * 128 + vc];

    // softmax over micro_logits[h, 0..3]
    float wmix[4];
    {
        float m0 = micro_logits[h * 4 + 0], m1 = micro_logits[h * 4 + 1];
        float m2 = micro_logits[h * 4 + 2], m3 = micro_logits[h * 4 + 3];
        float mx = fmaxf(fmaxf(m0, m1), fmaxf(m2, m3));
        wmix[0] = __expf(m0 - mx); wmix[1] = __expf(m1 - mx);
        wmix[2] = __expf(m2 - mx); wmix[3] = __expf(m3 - mx);
        float s = wmix[0] + wmix[1] + wmix[2] + wmix[3];
        float inv = 1.f / s;
        wmix[0] *= inv; wmix[1] *= inv; wmix[2] *= inv; wmix[3] *= inv;
    }

    float S[128];
    #pragma unroll
    for (int d = 0; d < 128; ++d) S[d] = 0.f;
    __syncthreads();

    for (int t = 0; t < 1024; ++t) {
        const size_t baseH = ((size_t)(b * 1024 + t) * 16 + h);
        const size_t qb = baseH * 128;        // (b,t,h,*) for q/decay/o
        const size_t kb = baseH * 512;        // (b,t,h,r,*) for k/v
        q_s[vc]   = q[qb + vc];
        dec_s[vc] = decay0[qb + vc];
        #pragma unroll
        for (int r = 0; r < 4; ++r) k_s[r][vc] = k[kb + r * 128 + vc];
        if (vc < 4) beta_s[vc] = beta[baseH * 4 + vc];
        __syncthreads();

        float o_acc = 0.f;
        #pragma unroll 1
        for (int r = 0; r < 4; ++r) {
            const float* dec = (r == 0) ? dec_s : dfill_s;
            const float* kk  = k_s[r];
            float u = 0.f;
            #pragma unroll
            for (int d = 0; d < 128; ++d) {
                float sv = S[d] * dec[d];
                S[d] = sv;
                u = fmaf(kk[d], sv, u);
            }
            float w = beta_s[r] * (v[kb + r * 128 + vc] - u);
            float outp = 0.f;
            #pragma unroll
            for (int d = 0; d < 128; ++d) {
                float sv = fmaf(kk[d], w, S[d]);
                S[d] = sv;
                outp = fmaf(q_s[d], sv, outp);
            }
            o_acc = fmaf(wmix[r], outp, o_acc);
        }
        o_mix[qb + vc] = o_acc;
        __syncthreads();
    }
}

// ---------------------------------------------------------------------------
// epilogue: RMS norm over DV + o_norm_w + sigmoid gate, -> bf16
// one wave32 per (b,t,h) group of 128.
// ---------------------------------------------------------------------------
__global__ __launch_bounds__(256)
void post_kernel(const float* __restrict__ o_mix,
                 const float* __restrict__ gateraw,
                 const float* __restrict__ g2_b,
                 const float* __restrict__ o_norm_w,
                 unsigned short* __restrict__ o_bf, int groups) {
    int wave = threadIdx.x >> 5;
    int lane = threadIdx.x & 31;
    int g = blockIdx.x * 8 + wave;
    if (g >= groups) return;
    int h = g & 15;
    size_t bt = (size_t)(g >> 4);
    const float* po = o_mix + (size_t)g * 128;
    float vals[4]; float s = 0.f;
    #pragma unroll
    for (int i = 0; i < 4; ++i) {
        vals[i] = po[lane + 32 * i];
        s += vals[i] * vals[i];
    }
    #pragma unroll
    for (int off = 16; off > 0; off >>= 1) s += __shfl_xor(s, off, 32);
    float inv = rsqrtf(s * (1.f / 128.f) + 1e-5f);
    #pragma unroll
    for (int i = 0; i < 4; ++i) {
        int dv = lane + 32 * i;
        int ch = h * 128 + dv;
        float gate = gateraw[bt * 2048 + ch] + g2_b[ch];
        float o = vals[i] * inv * o_norm_w[dv] * sigmoid_f(gate);
        o_bf[bt * 2048 + ch] = f32_to_bf16(o);
    }
}

// ---------------------------------------------------------------------------
// host side
// ---------------------------------------------------------------------------
extern "C" void kernel_launch(void* const* d_in, const int* in_sizes, int n_in,
                              void* d_out, int out_size, void* d_ws, size_t ws_size,
                              hipStream_t stream) {
    (void)in_sizes; (void)n_in; (void)out_size; (void)ws_size;

    const float* x_f       = (const float*)d_in[0];
    const float* q_proj_w  = (const float*)d_in[1];
    const float* k_proj_w  = (const float*)d_in[2];
    const float* v_proj_w  = (const float*)d_in[3];
    const float* q_conv_w  = (const float*)d_in[4];
    const float* k_conv_w  = (const float*)d_in[5];
    const float* v_conv_w  = (const float*)d_in[6];
    const float* f1_w      = (const float*)d_in[7];
    const float* f2_w      = (const float*)d_in[8];
    const float* b_proj_w  = (const float*)d_in[9];
    const float* micro_log = (const float*)d_in[10];
    const float* A_log     = (const float*)d_in[11];
    const float* dt_bias   = (const float*)d_in[12];
    const float* g1_w      = (const float*)d_in[13];
    const float* g2_w      = (const float*)d_in[14];
    const float* g2_b      = (const float*)d_in[15];
    const float* o_norm_w  = (const float*)d_in[16];
    const float* o_proj_w  = (const float*)d_in[17];
    float* out             = (float*)d_out;

    const int B = 2, T = 1024, HID = 2048;
    const int M = B * T;                 // 2048
    const long long MT = (long long)M;

    char* ws = (char*)d_ws;
    size_t off = 0;
    auto alloc = [&](size_t bytes) -> void* {
        void* p = ws + off;
        off += (bytes + 255) & ~(size_t)255;
        return p;
    };
    typedef unsigned short ushort_t;

    // bf16 buffers
    ushort_t* x_bf  = (ushort_t*)alloc((size_t)M * 2048 * 2);
    ushort_t* qw_bf = (ushort_t*)alloc((size_t)2048 * 2048 * 2);
    ushort_t* kw_bf = (ushort_t*)alloc((size_t)8192 * 2048 * 2);
    ushort_t* vw_bf = (ushort_t*)alloc((size_t)8192 * 2048 * 2);
    ushort_t* f1_bf = (ushort_t*)alloc((size_t)128 * 2048 * 2);
    ushort_t* f2_bf = (ushort_t*)alloc((size_t)2048 * 128 * 2);
    ushort_t* bw_bf = (ushort_t*)alloc((size_t)64 * 2048 * 2);
    ushort_t* g1_bf = (ushort_t*)alloc((size_t)128 * 2048 * 2);
    ushort_t* g2_bf = (ushort_t*)alloc((size_t)2048 * 128 * 2);
    ushort_t* ow_bf = (ushort_t*)alloc((size_t)2048 * 2048 * 2);
    ushort_t* h1_bf = (ushort_t*)alloc((size_t)M * 128 * 2);
    ushort_t* h2_bf = (ushort_t*)alloc((size_t)M * 128 * 2);
    ushort_t* o_bf  = (ushort_t*)alloc((size_t)M * 2048 * 2);
    // fp32 buffers
    float* q_pre   = (float*)alloc((size_t)M * 2048 * 4);
    float* k_pre   = (float*)alloc((size_t)M * 8192 * 4);
    float* v_pre   = (float*)alloc((size_t)M * 8192 * 4);
    float* q_post  = (float*)alloc((size_t)M * 2048 * 4);
    float* k_post  = (float*)alloc((size_t)M * 8192 * 4);
    float* v_post  = (float*)alloc((size_t)M * 8192 * 4);
    float* braw    = (float*)alloc((size_t)M * 64 * 4);
    float* h1      = (float*)alloc((size_t)M * 128 * 4);
    float* h2      = (float*)alloc((size_t)M * 128 * 4);
    float* graw    = (float*)alloc((size_t)M * 2048 * 4);   // -> decay0 in place
    float* gateraw = (float*)alloc((size_t)M * 2048 * 4);
    float* dfill   = (float*)alloc((size_t)16 * 128 * 4);
    float* o_mix   = (float*)alloc((size_t)M * 2048 * 4);

    auto cvt = [&](const float* src, ushort_t* dst, size_t n) {
        cvt_bf16_kernel<<<(unsigned)((n + 255) / 256), 256, 0, stream>>>(src, dst, n);
    };
    auto gemm = [&](const ushort_t* A, const ushort_t* W, float* C,
                    int Mm, int Nn, int Kk) {
        dim3 grid((Nn + 127) / 128, Mm / 128);
        gemm_bf16_kernel<<<grid, 256, 0, stream>>>(A, W, C, Mm, Nn, Kk);
    };

    // 1) conversions
    cvt(x_f,      x_bf,  (size_t)M * 2048);
    cvt(q_proj_w, qw_bf, (size_t)2048 * 2048);
    cvt(k_proj_w, kw_bf, (size_t)8192 * 2048);
    cvt(v_proj_w, vw_bf, (size_t)8192 * 2048);
    cvt(f1_w,     f1_bf, (size_t)128 * 2048);
    cvt(f2_w,     f2_bf, (size_t)2048 * 128);
    cvt(b_proj_w, bw_bf, (size_t)64 * 2048);
    cvt(g1_w,     g1_bf, (size_t)128 * 2048);
    cvt(g2_w,     g2_bf, (size_t)2048 * 128);
    cvt(o_proj_w, ow_bf, (size_t)2048 * 2048);

    // 2) projection GEMMs
    gemm(x_bf, qw_bf, q_pre, M, 2048, 2048);
    gemm(x_bf, kw_bf, k_pre, M, 8192, 2048);
    gemm(x_bf, vw_bf, v_pre, M, 8192, 2048);
    gemm(x_bf, bw_bf, braw,  M, 64,   2048);
    gemm(x_bf, f1_bf, h1,    M, 128,  2048);
    gemm(x_bf, g1_bf, h2,    M, 128,  2048);
    cvt(h1, h1_bf, (size_t)M * 128);
    cvt(h2, h2_bf, (size_t)M * 128);
    gemm(h1_bf, f2_bf, graw,    M, 2048, 128);
    gemm(h2_bf, g2_bf, gateraw, M, 2048, 128);

    // 3) conv + SiLU
    {
        long long nq = MT * 2048, nk = MT * 8192;
        conv_silu_kernel<<<(unsigned)((nq + 255) / 256), 256, 0, stream>>>(
            q_pre, q_conv_w, q_post, T, 2048, nq);
        conv_silu_kernel<<<(unsigned)((nk + 255) / 256), 256, 0, stream>>>(
            k_pre, k_conv_w, k_post, T, 8192, nk);
        conv_silu_kernel<<<(unsigned)((nk + 255) / 256), 256, 0, stream>>>(
            v_pre, v_conv_w, v_post, T, 8192, nk);
    }

    // 4) prep
    decay_fill_kernel<<<(2048 + 255) / 256, 256, 0, stream>>>(A_log, dt_bias, dfill, 2048);
    decay0_kernel<<<(unsigned)((MT * 2048 + 255) / 256), 256, 0, stream>>>(
        graw, A_log, dt_bias, MT * 2048);
    sigmoid_inplace_kernel<<<(unsigned)((MT * 64 + 255) / 256), 256, 0, stream>>>(
        braw, MT * 64);
    l2norm_kernel<<<(M * 16 + 7) / 8, 256, 0, stream>>>(
        q_post, M * 16, 0.08838834764831845f /* 1/sqrt(128) */);
    l2norm_kernel<<<(M * 16 * 4 + 7) / 8, 256, 0, stream>>>(
        k_post, M * 16 * 4, 1.0f);

    // 5) sequential scan: 32 blocks (B*H), 128 threads each
    scan_kernel<<<B * 16, 128, 0, stream>>>(
        q_post, k_post, v_post, braw, graw, dfill, micro_log, o_mix);

    // 6) epilogue -> bf16
    post_kernel<<<(M * 16 + 7) / 8, 256, 0, stream>>>(
        o_mix, gateraw, g2_b, o_norm_w, o_bf, M * 16);

    // 7) output projection
    gemm(o_bf, ow_bf, out, M, HID, 2048);
}
